// SegmentRewardModel_72610717106555
// MI455X (gfx1250) — matrix operand order repaired and checked by reading, not compiled, not run on hardware
//
#include <hip/hip_runtime.h>

typedef __attribute__((ext_vector_type(16))) _Float16 v16h;
typedef __attribute__((ext_vector_type(8)))  _Float16 v8h;
typedef __attribute__((ext_vector_type(8)))  float    v8f;
typedef __attribute__((ext_vector_type(4)))  unsigned int v4u;
typedef __attribute__((ext_vector_type(8)))  int v8i;
typedef __attribute__((ext_vector_type(4)))  int v4i;

#define SEG_B 256
#define SEG_T 2048
#define S_DIM 39
#define A_DIM 4
#define HID   256
#define ROWS_PER_TILE 128
#define TILES_PER_BLOCK 8
#define ROWS_PER_BLOCK (ROWS_PER_TILE * TILES_PER_BLOCK)   // 1024
#define LN_EPS 1e-5f

// LDS row strides (in halfs), padded for bank spread
#define SX_STRIDE  72
#define SW1_STRIDE 72
#define SW2_STRIDE 264
#define SH1_STRIDE 264

// byte offsets into dynamic LDS (all 16B aligned)
#define OFF_W1T   0                       // 256*72*2  = 36864
#define OFF_W2T   36864                   // 256*264*2 = 135168
#define OFF_X     172032                  // 128*72*2  = 18432
#define OFF_H1    190464                  // 128*264*2 = 67584
#define OFF_OBS0  258048                  // 128*39*4  = 19968
#define OFF_OBS1  278016                  // 19968
#define OFF_ACT0  297984                  // 128*4*4   = 2048
#define OFF_ACT1  300032                  // 2048
#define OFF_PAR   302080                  // 7*256*4   = 7168
#define OFF_RED   309248                  // 16*4
#define SMEM_BYTES 309312

#if defined(__HIP_DEVICE_COMPILE__) && !__has_builtin(__builtin_amdgcn_tensor_load_to_lds)
#error "no __builtin_amdgcn_tensor_load_to_lds on this toolchain"
#endif

union V16 { v16h v; v8h h[2]; };

__device__ __forceinline__ float hred16(float v) {
  v += __shfl_xor(v, 1, 32);
  v += __shfl_xor(v, 2, 32);
  v += __shfl_xor(v, 4, 32);
  v += __shfl_xor(v, 8, 32);
  return v;
}

// generic -> LDS(addrspace 3) byte offset
__device__ __forceinline__ unsigned lds_off(const void* p) {
  return (unsigned)(unsigned long long)
      (__attribute__((address_space(3))) const char*)(p);
}

// Issue one 2D TDM load: rows x dim0 elements, row stride stride0 (elements).
// data_size_code: 0=1B,1=2B,2=4B,3=8B. pad_bits: pad_enable/interval/amount
// pre-shifted for D# group1 word0 (bits 20, 24:22, 31:25).
__device__ __forceinline__ void tdm_load_2d(unsigned lds_byte_off,
                                            const void* gptr,
                                            unsigned dim0, unsigned rows,
                                            unsigned stride0,
                                            unsigned data_size_code,
                                            unsigned pad_bits) {
#if defined(__HIP_DEVICE_COMPILE__)
  unsigned long long ga = (unsigned long long)gptr;
  v4u g0;
  g0.x = 1u;                                             // count=1, user desc
  g0.y = lds_byte_off;                                   // lds_addr
  g0.z = (unsigned)(ga & 0xffffffffu);                   // global_addr[31:0]
  g0.w = (unsigned)((ga >> 32) & 0x01ffffffu)            // global_addr[56:32]
       | (2u << 30);                                     // type=2 (image)
  v8i g1;
  g1[0] = (int)((data_size_code << 16) | pad_bits);      // wg_mask=0, ds, pad
  g1[1] = (int)((dim0 & 0xffffu) << 16);                 // tensor_dim0[15:0]
  g1[2] = (int)(((dim0 >> 16) & 0xffffu)                 // tensor_dim0[31:16]
       | ((rows & 0xffffu) << 16));                      // tensor_dim1[15:0]
  g1[3] = (int)(((rows >> 16) & 0xffffu)                 // tensor_dim1[31:16]
       | ((dim0 & 0xffffu) << 16));                      // tile_dim0
  g1[4] = (int)(rows & 0xffffu);                         // tile_dim1, tile_dim2=0
  g1[5] = (int)stride0;                                  // dim0_stride[31:0]
  g1[6] = 0;                                             // stride hi, dim1_stride lo
  g1[7] = 0;
  v4i z4 = {0, 0, 0, 0};
  v8i z8 = {0, 0, 0, 0, 0, 0, 0, 0};
  __builtin_amdgcn_tensor_load_to_lds(g0, g1, z4, z4, z8, 0);
#endif
}

// ---- prep: transpose + fp32->fp16 weights into workspace ----
__global__ void srm_prep_weights(const float* __restrict__ W1,
                                 const float* __restrict__ W2,
                                 _Float16* __restrict__ w1t,   // [256][64]
                                 _Float16* __restrict__ w2t) { // [256][256]
  int idx = blockIdx.x * 256 + threadIdx.x;
  int stride = gridDim.x * 256;
  for (int e = idx; e < 256 * 64; e += stride) {
    int n = e >> 6, k = e & 63;
    w1t[e] = (k < (S_DIM + A_DIM)) ? (_Float16)W1[k * HID + n] : (_Float16)0.f;
  }
  for (int e = idx; e < 256 * 256; e += stride) {
    int n = e >> 8, k = e & 255;
    w2t[e] = (_Float16)W2[k * HID + n];
  }
}

__global__ void srm_zero_out(float* __restrict__ out) {
  out[threadIdx.x] = 0.f;
}

// ---- persistent fused MLP: TDM-fed, double-buffered row tiles ----
__global__ void __launch_bounds__(256, 1)
srm_fused(const float* __restrict__ obs, const float* __restrict__ act,
          const _Float16* __restrict__ w1t, const _Float16* __restrict__ w2t,
          const float* __restrict__ b1, const float* __restrict__ g1, const float* __restrict__ bt1,
          const float* __restrict__ b2, const float* __restrict__ g2, const float* __restrict__ bt2,
          const float* __restrict__ W3, const float* __restrict__ b3,
          float* __restrict__ out)
{
  extern __shared__ char sm[];
  _Float16* sW1t = (_Float16*)(sm + OFF_W1T);
  _Float16* sW2t = (_Float16*)(sm + OFF_W2T);
  _Float16* sX   = (_Float16*)(sm + OFF_X);
  _Float16* sH1  = (_Float16*)(sm + OFF_H1);
  float* sObs0 = (float*)(sm + OFF_OBS0);
  float* sObs1 = (float*)(sm + OFF_OBS1);
  float* sAct0 = (float*)(sm + OFF_ACT0);
  float* sAct1 = (float*)(sm + OFF_ACT1);
  float* sB1  = (float*)(sm + OFF_PAR);
  float* sG1  = sB1 + 256;  float* sBt1 = sB1 + 512;
  float* sB2  = sB1 + 768;  float* sG2  = sB1 + 1024; float* sBt2 = sB1 + 1280;
  float* sW3  = sB1 + 1536;
  float* sRed = (float*)(sm + OFF_RED);

  const int tid      = threadIdx.x;
  const int rowBase  = blockIdx.x * ROWS_PER_BLOCK;
  const int lane = tid & 31, wave = tid >> 5;
  const int lo = lane & 15, hi = lane >> 4;
  const int m0 = wave * 16;                 // wave-private 16-row tile

  const unsigned smBase = lds_off(sm);      // LDS byte offset of dynamic base

  // ---- prologue: TDM weights (padded into LDS) + row-tile 0 ----
  if (wave == 0) {
    // W1t: 256 rows x 64 halfs, pad +4 dwords per 32-dword row (stride 72h)
    tdm_load_2d(smBase + OFF_W1T, w1t, 64, 256, 64, 1u,
                (1u << 20) | (4u << 22) | (3u << 25));
    // W2t: 256 rows x 256 halfs, pad +4 dwords per 128-dword row (stride 264h)
    tdm_load_2d(smBase + OFF_W2T, w2t, 256, 256, 256, 1u,
                (1u << 20) | (6u << 22) | (3u << 25));
    // tile 0 staging: raw fp32 rows
    tdm_load_2d(smBase + OFF_OBS0, obs + (long long)rowBase * S_DIM,
                S_DIM, ROWS_PER_TILE, S_DIM, 2u, 0u);
    tdm_load_2d(smBase + OFF_ACT0, act + (long long)rowBase * A_DIM,
                A_DIM, ROWS_PER_TILE, A_DIM, 2u, 0u);
  }
  // small params via normal loads (uniform, L2-hot)
  sB1[tid] = b1[tid]; sG1[tid] = g1[tid]; sBt1[tid] = bt1[tid];
  sB2[tid] = b2[tid]; sG2[tid] = g2[tid]; sBt2[tid] = bt2[tid];
  sW3[tid] = W3[tid];

  float blockAcc = 0.f;
  v8f acc[16];

  for (int t = 0; t < TILES_PER_BLOCK; ++t) {
    const int buf = t & 1;
    // ---- issue next tile's DMA, then wait for current tile (+weights) ----
    if (wave == 0) {
      if (t + 1 < TILES_PER_BLOCK) {
        int nrow = rowBase + (t + 1) * ROWS_PER_TILE;
        int nbuf = (t + 1) & 1;
        tdm_load_2d(smBase + (nbuf ? OFF_OBS1 : OFF_OBS0),
                    obs + (long long)nrow * S_DIM,
                    S_DIM, ROWS_PER_TILE, S_DIM, 2u, 0u);
        tdm_load_2d(smBase + (nbuf ? OFF_ACT1 : OFF_ACT0),
                    act + (long long)nrow * A_DIM,
                    A_DIM, ROWS_PER_TILE, A_DIM, 2u, 0u);
        __builtin_amdgcn_s_wait_tensorcnt(2);   // current tile's pair is done
      } else {
        __builtin_amdgcn_s_wait_tensorcnt(0);
      }
    }
    __syncthreads();   // publish TDM-written LDS to all waves

    // ---- convert staging f32 -> sX f16 (concat + pad K 43->64) ----
    {
      const float* ob = buf ? sObs1 : sObs0;
      const float* ac = buf ? sAct1 : sAct0;
      for (int e = tid; e < ROWS_PER_TILE * 64; e += 256) {
        int r = e >> 6, k = e & 63;
        float v = 0.f;
        if (k < S_DIM)              v = ob[r * S_DIM + k];
        else if (k < S_DIM + A_DIM) v = ac[r * A_DIM + (k - S_DIM)];
        sX[r * SX_STRIDE + k] = (_Float16)v;
      }
    }
    __syncthreads();

    // ================= Layer 1: [16x64] x [64x256] =================
#pragma unroll
    for (int q = 0; q < 16; ++q) acc[q] = (v8f){0,0,0,0,0,0,0,0};
#pragma unroll
    for (int kc = 0; kc < 2; ++kc) {
      V16 a;
      const _Float16* ap = sX + (m0 + lo) * SX_STRIDE + kc * 32 + hi * 8;
      a.h[0] = *(const v8h*)ap;
      a.h[1] = *(const v8h*)(ap + 16);
#pragma unroll
      for (int q = 0; q < 16; ++q) {
        V16 bfrag;
        const _Float16* bp = sW1t + (q * 16 + lo) * SW1_STRIDE + kc * 32 + hi * 8;
        bfrag.h[0] = *(const v8h*)bp;
        bfrag.h[1] = *(const v8h*)(bp + 16);
        acc[q] = __builtin_amdgcn_wmma_f32_16x16x32_f16(
            false, a.v, false, bfrag.v, (short)0, acc[q], false, false);
      }
    }

    // ---- LN1 + ReLU in C-fragment layout; stage h1 (f16) ----
#pragma unroll
    for (int j = 0; j < 8; ++j) {
      float s = 0.f, s2 = 0.f;
#pragma unroll
      for (int q = 0; q < 16; ++q) {
        float v = acc[q][j] + sB1[q * 16 + lo];
        acc[q][j] = v;
        s += v; s2 += v * v;
      }
      s = hred16(s); s2 = hred16(s2);
      float mean = s * (1.f / 256.f);
      float var  = s2 * (1.f / 256.f) - mean * mean;
      float rinv = rsqrtf(var + LN_EPS);
      int r = m0 + j + hi * 8;
#pragma unroll
      for (int q = 0; q < 16; ++q) {
        int col = q * 16 + lo;
        float v = (acc[q][j] - mean) * rinv * sG1[col] + sBt1[col];
        v = fmaxf(v, 0.f);
        sH1[r * SH1_STRIDE + col] = (_Float16)v;   // same-wave LDS: in-order
      }
    }

    // ================= Layer 2: [16x256] x [256x256] =================
#pragma unroll
    for (int q = 0; q < 16; ++q) acc[q] = (v8f){0,0,0,0,0,0,0,0};
#pragma unroll
    for (int kc = 0; kc < 8; ++kc) {
      V16 a;
      const _Float16* ap = sH1 + (m0 + lo) * SH1_STRIDE + kc * 32 + hi * 8;
      a.h[0] = *(const v8h*)ap;
      a.h[1] = *(const v8h*)(ap + 16);
#pragma unroll
      for (int q = 0; q < 16; ++q) {
        V16 bfrag;
        const _Float16* bp = sW2t + (q * 16 + lo) * SW2_STRIDE + kc * 32 + hi * 8;
        bfrag.h[0] = *(const v8h*)bp;
        bfrag.h[1] = *(const v8h*)(bp + 16);
        acc[q] = __builtin_amdgcn_wmma_f32_16x16x32_f16(
            false, a.v, false, bfrag.v, (short)0, acc[q], false, false);
      }
    }

    // ---- LN2 + ReLU + dot(W3) + per-wave row-reward sum ----
    float wsum = 0.f;
#pragma unroll
    for (int j = 0; j < 8; ++j) {
      float s = 0.f, s2 = 0.f;
#pragma unroll
      for (int q = 0; q < 16; ++q) {
        float v = acc[q][j] + sB2[q * 16 + lo];
        acc[q][j] = v;
        s += v; s2 += v * v;
      }
      s = hred16(s); s2 = hred16(s2);
      float mean = s * (1.f / 256.f);
      float var  = s2 * (1.f / 256.f) - mean * mean;
      float rinv = rsqrtf(var + LN_EPS);
      float d = 0.f;
#pragma unroll
      for (int q = 0; q < 16; ++q) {
        int col = q * 16 + lo;
        float v = (acc[q][j] - mean) * rinv * sG2[col] + sBt2[col];
        v = fmaxf(v, 0.f);
        d += v * sW3[col];
      }
      d = hred16(d);
      wsum += d;
    }
    if (lo == 0) sRed[wave * 2 + hi] = wsum;
    __syncthreads();
    if (tid == 0) {
      float tot = 0.f;
      for (int i = 0; i < 16; ++i) tot += sRed[i];
      blockAcc += tot;
    }
  }

  if (tid == 0) {
    blockAcc += (float)ROWS_PER_BLOCK * b3[0];
    atomicAdd(&out[blockIdx.x >> 1], blockAcc);  // 2 blocks per batch (2048/1024)
  }
}

extern "C" void kernel_launch(void* const* d_in, const int* in_sizes, int n_in,
                              void* d_out, int out_size, void* d_ws, size_t ws_size,
                              hipStream_t stream) {
  const float* obs = (const float*)d_in[0];
  const float* act = (const float*)d_in[1];
  const float* W1  = (const float*)d_in[2];
  const float* b1  = (const float*)d_in[3];
  const float* g1  = (const float*)d_in[4];
  const float* bt1 = (const float*)d_in[5];
  const float* W2  = (const float*)d_in[6];
  const float* b2  = (const float*)d_in[7];
  const float* g2  = (const float*)d_in[8];
  const float* bt2 = (const float*)d_in[9];
  const float* W3  = (const float*)d_in[10];
  const float* b3  = (const float*)d_in[11];
  float* out = (float*)d_out;

  _Float16* w1t = (_Float16*)d_ws;            // 256*64 halfs = 32 KB
  _Float16* w2t = w1t + 256 * 64;             // 256*256 halfs = 128 KB

  srm_zero_out<<<1, 256, 0, stream>>>(out);
  srm_prep_weights<<<64, 256, 0, stream>>>(W1, W2, w1t, w2t);

  int grid = (SEG_B * SEG_T) / ROWS_PER_BLOCK; // 512 persistent blocks
  srm_fused<<<grid, 256, SMEM_BYTES, stream>>>(
      obs, act, w1t, w2t, b1, g1, bt1, b2, g2, bt2, W3, b3, out);
}
